// SelfAttention_58093727646333
// MI455X (gfx1250) — matrix operand order repaired and checked by reading, not compiled, not run on hardware
//
#include <hip/hip_runtime.h>
#include <hip/hip_bf16.h>

#define H_   16
#define D_   64
#define S_   2048
#define N_   4
#define TOK  (N_ * S_)          // 8192 tokens

typedef __attribute__((ext_vector_type(16))) _Float16 v16h;
typedef __attribute__((ext_vector_type(8)))  float    v8f;

static __device__ __forceinline__ v8f wmma16(v16h a, v16h b, v8f c) {
    // D = A(16x32 f16) * B(32x16 f16) + C(16x16 f32)
    return __builtin_amdgcn_wmma_f32_16x16x32_f16(
        /*neg_a=*/false, a, /*neg_b=*/false, b,
        /*c_mod=*/(short)0, c, /*reuse_a=*/false, /*reuse_b=*/false);
}

// ---------------------------------------------------------------------------
// Kernel 1: per-head projection  P = X[:, h*64 : h*64+64] @ W[h] + b[h]
// X: [TOK][1024] f32.  W: [H][64][64] f32 (out[e] = sum_d x[d]*W[d][e]).
// transposed==0 : out f16 [n][h][s][64]   (Q-hat / K-hat)
// transposed==1 : out f16 [n][h][64][S]   (V-hat, transposed for AV B-tiles)
// grid = (TOK/64, H), block = 128 (4 waves, 16 rows each)
// ---------------------------------------------------------------------------
__global__ __launch_bounds__(128)
void proj_head_kernel(const float* __restrict__ X, const float* __restrict__ W,
                      const float* __restrict__ bias, _Float16* __restrict__ out,
                      int transposed) {
    const int lane = threadIdx.x & 31;
    const int wave = threadIdx.x >> 5;
    const int lrow = lane & 15;     // M (A) or N (B/C) index
    const int lhi  = lane >> 4;     // lane half
    const int h    = blockIdx.y;
    const int t0   = blockIdx.x * 64 + wave * 16;

    v8f acc[4];
    #pragma unroll
    for (int nt = 0; nt < 4; ++nt)
        #pragma unroll
        for (int i = 0; i < 8; ++i) acc[nt][i] = 0.0f;

    const float* xrow = X + (size_t)(t0 + lrow) * 1024 + h * 64;
    const float* Wh   = W + h * 64 * 64;

    #pragma unroll
    for (int kk = 0; kk < 64; kk += 32) {
        v16h a;
        #pragma unroll
        for (int e = 0; e < 8; ++e) a[e]     = (_Float16)xrow[kk + lhi * 16 + e];
        #pragma unroll
        for (int e = 0; e < 8; ++e) a[8 + e] = (_Float16)xrow[kk + lhi * 16 + 8 + e];
        #pragma unroll
        for (int nt = 0; nt < 4; ++nt) {
            v16h b;
            #pragma unroll
            for (int e = 0; e < 16; ++e)
                b[e] = (_Float16)Wh[(kk + lhi * 16 + e) * 64 + nt * 16 + lrow];
            acc[nt] = wmma16(a, b, acc[nt]);
        }
    }

    #pragma unroll
    for (int nt = 0; nt < 4; ++nt) {
        const float bb = bias[h * 64 + nt * 16 + lrow];
        const int   d  = nt * 16 + lrow;
        #pragma unroll
        for (int r = 0; r < 8; ++r) {
            float v = acc[nt][r] + bb;
            int g  = t0 + r + lhi * 8;          // global token
            int nn = g >> 11;                   // / 2048
            int ss = g & (S_ - 1);
            size_t idx = transposed
                ? ((size_t)(nn * H_ + h) * 64 + d) * S_ + ss
                : ((size_t)(nn * H_ + h) * S_ + ss) * 64 + d;
            out[idx] = (_Float16)v;
        }
    }
}

// ---------------------------------------------------------------------------
// Kernel 2: Wo f32 -> f16 (once)
// ---------------------------------------------------------------------------
__global__ __launch_bounds__(256)
void cvt_wo_kernel(const float* __restrict__ Wo, _Float16* __restrict__ Wo16) {
    int i = blockIdx.x * 256 + threadIdx.x;
    Wo16[i] = (_Float16)Wo[i];
}

// ---------------------------------------------------------------------------
// Kernel 3: flash attention per (n,h).
// Qp/Kp: f16 [n][h][S][64], Vt: f16 [n][h][64][S], mask: int [N][S].
// O: f16 [n][q][1024] attention output (feature = h*64+d).
// grid = (S/256, N*H), block = 256 (8 waves, 32 queries/wave)
// ---------------------------------------------------------------------------
__global__ __launch_bounds__(256)
void flash_attn_kernel(const _Float16* __restrict__ Qp,
                       const _Float16* __restrict__ Kp,
                       const _Float16* __restrict__ Vt,
                       const int* __restrict__ mask,
                       _Float16* __restrict__ O) {
    __shared__ __align__(32) _Float16 ldsP[8 * 32 * 32];   // 16 KB, 2KB/wave

    const int lane = threadIdx.x & 31;
    const int wave = threadIdx.x >> 5;
    const int lrow = lane & 15;
    const int lhi  = lane >> 4;
    const int nh   = blockIdx.y;
    const int n    = nh >> 4;
    const int h    = nh & 15;
    const int q0   = blockIdx.x * 256 + wave * 32;

    const _Float16* Qb = Qp + (size_t)nh * S_ * 64;
    const _Float16* Kb = Kp + (size_t)nh * S_ * 64;
    const _Float16* Vb = Vt + (size_t)nh * 64 * S_;
    const int* mrow = mask + n * S_;
    _Float16* lds = ldsP + wave * 32 * 32;

    // cache Q A-tiles: [mt][kk]  (32 queries x 64 dim)
    v16h aQ[2][2];
    #pragma unroll
    for (int mt = 0; mt < 2; ++mt) {
        const _Float16* qrow = Qb + (size_t)(q0 + mt * 16 + lrow) * 64;
        #pragma unroll
        for (int kk = 0; kk < 2; ++kk) {
            #pragma unroll
            for (int e = 0; e < 8; ++e) aQ[mt][kk][e]     = qrow[kk * 32 + lhi * 16 + e];
            #pragma unroll
            for (int e = 0; e < 8; ++e) aQ[mt][kk][8 + e] = qrow[kk * 32 + lhi * 16 + 8 + e];
        }
    }

    float mstate[2][8], lstate[2][8];
    v8f   o[2][4];
    #pragma unroll
    for (int mt = 0; mt < 2; ++mt) {
        #pragma unroll
        for (int r = 0; r < 8; ++r) { mstate[mt][r] = -3.0e38f; lstate[mt][r] = 0.0f; }
        #pragma unroll
        for (int nt = 0; nt < 4; ++nt)
            #pragma unroll
            for (int r = 0; r < 8; ++r) o[mt][nt][r] = 0.0f;
    }

    const float scale = 0.03125f;   // 1/sqrt(VOXEL_DIM) = 1/32

    for (int jb = 0; jb < S_; jb += 32) {
        // ---- scores S = Q K^T (32q x 32k), K B-tiles: b[K=d][N=j] = Kp[j][d]
        v16h bK[2][2];
        #pragma unroll
        for (int kt = 0; kt < 2; ++kt) {
            const _Float16* krow = Kb + (size_t)(jb + kt * 16 + lrow) * 64;
            #pragma unroll
            for (int kk = 0; kk < 2; ++kk)
                bK[kt][kk] = *(const v16h*)(krow + kk * 32 + lhi * 16);
        }
        v8f s[2][2];
        #pragma unroll
        for (int mt = 0; mt < 2; ++mt)
            #pragma unroll
            for (int kt = 0; kt < 2; ++kt) {
                v8f z;
                #pragma unroll
                for (int r = 0; r < 8; ++r) z[r] = 0.0f;
                z = wmma16(aQ[mt][0], bK[kt][0], z);
                z = wmma16(aQ[mt][1], bK[kt][1], z);
                s[mt][kt] = z;
            }

        // ---- mask + scale (column per lane)
        int mv[2];
        #pragma unroll
        for (int kt = 0; kt < 2; ++kt) mv[kt] = mrow[jb + kt * 16 + lrow];
        #pragma unroll
        for (int mt = 0; mt < 2; ++mt)
            #pragma unroll
            for (int kt = 0; kt < 2; ++kt)
                #pragma unroll
                for (int r = 0; r < 8; ++r)
                    s[mt][kt][r] = mv[kt] ? s[mt][kt][r] * scale : -1.0e30f;

        // ---- online softmax: row max (reduce over 16 lanes of each half)
        float alpha[2][8];
        #pragma unroll
        for (int mt = 0; mt < 2; ++mt)
            #pragma unroll
            for (int r = 0; r < 8; ++r) {
                float t = fmaxf(s[mt][0][r], s[mt][1][r]);
                t = fmaxf(t, __shfl_xor(t, 1, 32));
                t = fmaxf(t, __shfl_xor(t, 2, 32));
                t = fmaxf(t, __shfl_xor(t, 4, 32));
                t = fmaxf(t, __shfl_xor(t, 8, 32));
                float mn = fmaxf(mstate[mt][r], t);
                alpha[mt][r] = __expf(mstate[mt][r] - mn);
                mstate[mt][r] = mn;
            }

        // ---- P = exp(S - m), stage to LDS in row-major [32q][32k] f16
        #pragma unroll
        for (int mt = 0; mt < 2; ++mt)
            #pragma unroll
            for (int kt = 0; kt < 2; ++kt)
                #pragma unroll
                for (int r = 0; r < 8; ++r) {
                    float p = __expf(s[mt][kt][r] - mstate[mt][r]);
                    s[mt][kt][r] = p;
                    lds[(mt * 16 + r + lhi * 8) * 32 + kt * 16 + lrow] = (_Float16)p;
                }

        // ---- row sums, update l, rescale O
        #pragma unroll
        for (int mt = 0; mt < 2; ++mt)
            #pragma unroll
            for (int r = 0; r < 8; ++r) {
                float rs = s[mt][0][r] + s[mt][1][r];
                rs += __shfl_xor(rs, 1, 32);
                rs += __shfl_xor(rs, 2, 32);
                rs += __shfl_xor(rs, 4, 32);
                rs += __shfl_xor(rs, 8, 32);
                lstate[mt][r] = lstate[mt][r] * alpha[mt][r] + rs;
            }
        #pragma unroll
        for (int mt = 0; mt < 2; ++mt)
            #pragma unroll
            for (int nt = 0; nt < 4; ++nt)
                #pragma unroll
                for (int r = 0; r < 8; ++r) o[mt][nt][r] *= alpha[mt][r];

        // ---- O += P @ V : B[K=j][N=d] = Vt[d][j] (contiguous along j)
        v16h bV[4];
        #pragma unroll
        for (int nt = 0; nt < 4; ++nt)
            bV[nt] = *(const v16h*)(Vb + (size_t)(nt * 16 + lrow) * S_ + jb + lhi * 16);
        #pragma unroll
        for (int mt = 0; mt < 2; ++mt) {
            const _Float16* prow = lds + (mt * 16 + lrow) * 32;
            v16h aP;
            #pragma unroll
            for (int e = 0; e < 8; ++e) aP[e]     = prow[lhi * 16 + e];
            #pragma unroll
            for (int e = 0; e < 8; ++e) aP[8 + e] = prow[lhi * 16 + 8 + e];
            #pragma unroll
            for (int nt = 0; nt < 4; ++nt)
                o[mt][nt] = wmma16(aP, bV[nt], o[mt][nt]);
        }
    }

    // ---- finalize: O /= l, store f16 to [n][q][h*64+d]
    #pragma unroll
    for (int mt = 0; mt < 2; ++mt) {
        float inv[8];
        #pragma unroll
        for (int r = 0; r < 8; ++r)
            inv[r] = (lstate[mt][r] > 0.0f) ? 1.0f / lstate[mt][r] : 0.0f;
        #pragma unroll
        for (int nt = 0; nt < 4; ++nt) {
            const int d = nt * 16 + lrow;
            #pragma unroll
            for (int r = 0; r < 8; ++r) {
                int q = q0 + mt * 16 + r + lhi * 8;
                O[((size_t)(n * S_ + q)) * 1024 + h * 64 + d] =
                    (_Float16)(o[mt][nt][r] * inv[r]);
            }
        }
    }
}

// ---------------------------------------------------------------------------
// Kernel 4: Y = AO @ Wo^T + bo.  AO f16 [8192][1024], Wo16 f16 [1024][1024]
// B[K=c][N=o] = Wo16[o][c]  (contiguous along c).  Y f32 [8192][1024].
// grid = (TOK/256, 1024/64), block = 256 (8 waves, 32 rows x 64 cols each)
// ---------------------------------------------------------------------------
__global__ __launch_bounds__(256)
void out_proj_kernel(const _Float16* __restrict__ A, const _Float16* __restrict__ Wo16,
                     const float* __restrict__ bo, float* __restrict__ Y) {
    const int lane = threadIdx.x & 31;
    const int wave = threadIdx.x >> 5;
    const int lrow = lane & 15;
    const int lhi  = lane >> 4;
    const int r0   = blockIdx.x * 256 + wave * 32;
    const int c0   = blockIdx.y * 64;

    v8f o[2][4];
    #pragma unroll
    for (int mt = 0; mt < 2; ++mt)
        #pragma unroll
        for (int nt = 0; nt < 4; ++nt)
            #pragma unroll
            for (int r = 0; r < 8; ++r) o[mt][nt][r] = 0.0f;

    for (int k0 = 0; k0 < 1024; k0 += 32) {
        v16h a[2];
        #pragma unroll
        for (int mt = 0; mt < 2; ++mt) {
            const _Float16* arow = A + (size_t)(r0 + mt * 16 + lrow) * 1024 + k0;
            #pragma unroll
            for (int e = 0; e < 8; ++e) a[mt][e]     = arow[lhi * 16 + e];
            #pragma unroll
            for (int e = 0; e < 8; ++e) a[mt][8 + e] = arow[lhi * 16 + 8 + e];
        }
        #pragma unroll
        for (int nt = 0; nt < 4; ++nt) {
            v16h b = *(const v16h*)(Wo16 + (size_t)(c0 + nt * 16 + lrow) * 1024 + k0 + lhi * 16);
            #pragma unroll
            for (int mt = 0; mt < 2; ++mt)
                o[mt][nt] = wmma16(a[mt], b, o[mt][nt]);
        }
    }

    #pragma unroll
    for (int nt = 0; nt < 4; ++nt) {
        const float bb = bo[c0 + nt * 16 + lrow];
        #pragma unroll
        for (int mt = 0; mt < 2; ++mt)
            #pragma unroll
            for (int r = 0; r < 8; ++r)
                Y[(size_t)(r0 + mt * 16 + r + lhi * 8) * 1024 + c0 + nt * 16 + lrow] =
                    o[mt][nt][r] + bb;
    }
}

// ---------------------------------------------------------------------------
extern "C" void kernel_launch(void* const* d_in, const int* in_sizes, int n_in,
                              void* d_out, int out_size, void* d_ws, size_t ws_size,
                              hipStream_t stream) {
    (void)in_sizes; (void)n_in; (void)out_size; (void)ws_size;
    const float* values = (const float*)d_in[0];
    const float* keys   = (const float*)d_in[1];
    const float* query  = (const float*)d_in[2];
    const int*   mask   = (const int*)  d_in[3];
    const float* Wv     = (const float*)d_in[4];
    const float* bv     = (const float*)d_in[5];
    const float* Wk     = (const float*)d_in[6];
    const float* bk     = (const float*)d_in[7];
    const float* Wq     = (const float*)d_in[8];
    const float* bq     = (const float*)d_in[9];
    const float* Wo     = (const float*)d_in[10];
    const float* bo     = (const float*)d_in[11];
    float* out = (float*)d_out;

    const size_t SZ = (size_t)N_ * H_ * S_ * 64;   // 8,388,608 halves (16 MB)
    _Float16* Qp   = (_Float16*)d_ws;
    _Float16* Kp   = Qp + SZ;
    _Float16* Vt   = Kp + SZ;
    _Float16* AO   = Vt + SZ;
    _Float16* Wo16 = AO + SZ;

    dim3 pgrid(TOK / 64, H_);
    proj_head_kernel<<<pgrid, 128, 0, stream>>>(query,  Wq, bq, Qp, 0);
    proj_head_kernel<<<pgrid, 128, 0, stream>>>(keys,   Wk, bk, Kp, 0);
    proj_head_kernel<<<pgrid, 128, 0, stream>>>(values, Wv, bv, Vt, 1);
    cvt_wo_kernel<<<(1024 * 1024) / 256, 256, 0, stream>>>(Wo, Wo16);
    flash_attn_kernel<<<dim3(S_ / 256, N_ * H_), 256, 0, stream>>>(Qp, Kp, Vt, mask, AO);
    out_proj_kernel<<<dim3(TOK / 256, 1024 / 64), 256, 0, stream>>>(AO, Wo16, bo, out);
}
